// ModelWrapper_49100066128303
// MI455X (gfx1250) — matrix operand order repaired and checked by reading, not compiled, not run on hardware
//
#include <hip/hip_runtime.h>
#include <math.h>

typedef __attribute__((ext_vector_type(16))) _Float16 v16h;
typedef __attribute__((ext_vector_type(8)))  float    v8f;

#define DIM_IN   128
#define XDIM     256              // 2*DIM_IN (concat width, GEMM K)
#define HIDDEN   1024
#define NT_N     (HIDDEN / 16)    // 64 N-tiles
#define KSTEPS   (XDIM / 32)      // 8 K-steps of 32

// ---------------------------------------------------------------------------
// GELU via gfx1250's native v_tanh_f32: 0.5*h*(1+tanh(0.79788456*(h+0.044715*h^3))).
// ~6 straight-line VALU ops per value (vs ~14 for the A&S erf path); deviation
// from exact-erf GELU (<~3e-4) is below the f16 WMMA quantization noise.
// ---------------------------------------------------------------------------
__device__ __forceinline__ float gelu_fast(float h) {
    float u     = h * h;
    float inner = h * fmaf(0.035677408136f, u, 0.797884560803f);
    float th    = __builtin_amdgcn_tanhf(inner);
    float half_h = 0.5f * h;
    return fmaf(half_h, th, half_h);
}

// ---------------------------------------------------------------------------
// Prep: W1 [256 x 1024] fp32 row-major  ->  f16 WMMA B-fragments in d_ws.
// Fragment (n,k) for lane l is a v16h at B[(n*KSTEPS + k)*32 + l].
// B (32x16, KxN) 16-bit layout: lanes 0-15 hold K=0..7,16..23, lanes 16-31 hold
// K=8..15,24..31 (2 K per VGPR), column N = n*16 + (l&15).
// ---------------------------------------------------------------------------
__global__ void prep_w1_kernel(const float* __restrict__ W1,
                               v16h* __restrict__ B) {
    int t = blockIdx.x * blockDim.x + threadIdx.x;
    if (t >= NT_N * KSTEPS * 32) return;
    int lane = t & 31;
    int k    = (t >> 5) & (KSTEPS - 1);
    int n    = t >> 8;                       // t / (32*KSTEPS)
    int khi  = (lane >= 16) ? 8 : 0;
    int N    = n * 16 + (lane & 15);
    v16h f;
#pragma unroll
    for (int e = 0; e < 16; ++e) {
        int K = k * 32 + khi + (e < 8 ? e : e + 8);
        f[e] = (_Float16)W1[K * HIDDEN + N];
    }
    B[t] = f;
}

// ---------------------------------------------------------------------------
// Fused gather + Linear(256->1024) + GELU + Linear(1024->2).
// One wave per 16-row tile of pairs.
// ---------------------------------------------------------------------------
__global__ void __launch_bounds__(256) pair_mlp_kernel(
    const float* __restrict__ emb,
    const int*   __restrict__ idxg,
    const int*   __restrict__ idxd,
    const v16h*  __restrict__ B,      // prepped W1 fragments
    const float* __restrict__ b1,
    const float* __restrict__ W2,     // [1024, 2] row-major
    const float* __restrict__ b2,
    float*       __restrict__ out,    // [P, 2]
    int npairs)
{
    const int lane = threadIdx.x & 31;
    const int wave = threadIdx.x >> 5;
    const int tile = blockIdx.x * (blockDim.x >> 5) + wave;
    const int row0 = tile * 16;
    if (row0 >= npairs) return;            // wave-uniform: EXEC stays all-1s

    const int m   = lane & 15;             // this lane's M row within tile
    int row = row0 + m;
    if (row >= npairs) row = npairs - 1;   // clamp keeps every lane active

    // ---- Gather + convert A fragments (16-bit A 16x32 layout), keep all
    // ---- K=256 resident: 8 fragments = 64 VGPRs.
    const int gi = idxg[row];
    const int di = idxd[row];
    const float* eg = emb + (size_t)gi * DIM_IN;
    const float* ed = emb + (size_t)di * DIM_IN;
    const int off = (lane >= 16) ? 8 : 0;  // lane-half K interleave

    v16h afrag[KSTEPS];
#pragma unroll
    for (int k = 0; k < KSTEPS; ++k) {
        int K1 = k * 32 + off;             // elements 0..7  -> K1..K1+7
        int K2 = K1 + 16;                  // elements 8..15 -> K2..K2+7
        // 8-float runs are 32B-aligned and never cross the 128 boundary.
        const float4* p1 = (const float4*)((K1 < DIM_IN) ? (eg + K1)
                                                         : (ed + (K1 - DIM_IN)));
        const float4* p2 = (const float4*)((K2 < DIM_IN) ? (eg + K2)
                                                         : (ed + (K2 - DIM_IN)));
        float4 x0 = p1[0], x1 = p1[1];
        float4 y0 = p2[0], y1 = p2[1];
        v16h f;
        f[0]  = (_Float16)x0.x; f[1]  = (_Float16)x0.y;
        f[2]  = (_Float16)x0.z; f[3]  = (_Float16)x0.w;
        f[4]  = (_Float16)x1.x; f[5]  = (_Float16)x1.y;
        f[6]  = (_Float16)x1.z; f[7]  = (_Float16)x1.w;
        f[8]  = (_Float16)y0.x; f[9]  = (_Float16)y0.y;
        f[10] = (_Float16)y0.z; f[11] = (_Float16)y0.w;
        f[12] = (_Float16)y1.x; f[13] = (_Float16)y1.y;
        f[14] = (_Float16)y1.z; f[15] = (_Float16)y1.w;
        afrag[k] = f;
    }

    // Per-lane layer-2 partial sums (this lane's N column, all 8 M rows).
    float o0[8], o1[8];
#pragma unroll
    for (int r = 0; r < 8; ++r) { o0[r] = 0.0f; o1[r] = 0.0f; }

    const v16h* bp = B + lane;
    const float2* W2v = (const float2*)W2;

    for (int n = 0; n < NT_N; ++n) {
        v8f c = {};
#pragma unroll
        for (int k = 0; k < KSTEPS; ++k) {
            v16h bf = bp[(n * KSTEPS + k) * 32];
            c = __builtin_amdgcn_wmma_f32_16x16x32_f16(
                    false, afrag[k], false, bf, (short)0, c, false, false);
        }
        // C/D layout: VGPR r -> M = r (lanes 0-15) or 8+r (lanes 16-31),
        // N = n*16 + (lane&15) for every r -> one bias / one W2 row per lane.
        int    ncol = n * 16 + m;
        float  b1v  = b1[ncol];
        float2 w2   = W2v[ncol];
#pragma unroll
        for (int r = 0; r < 8; ++r) {
            float g = gelu_fast(c[r] + b1v);
            o0[r] += g * w2.x;
            o1[r] += g * w2.y;
        }
    }

    // Reduce over the 16 N-lanes of each half-wave (butterfly stays in-group).
#pragma unroll
    for (int r = 0; r < 8; ++r) {
#pragma unroll
        for (int s = 8; s >= 1; s >>= 1) {
            o0[r] += __shfl_xor(o0[r], s, 32);
            o1[r] += __shfl_xor(o1[r], s, 32);
        }
    }

    if (m == 0) {                          // lane 0 -> rows 0..7, lane 16 -> 8..15
        const float bo0 = b2[0], bo1 = b2[1];
        const int mbase = row0 + ((lane >= 16) ? 8 : 0);
        float2* o = (float2*)out;
#pragma unroll
        for (int r = 0; r < 8; ++r) {
            int rr = mbase + r;
            if (rr < npairs)
                o[rr] = make_float2(o0[r] + bo0, o1[r] + bo1);
        }
    }
}

// ---------------------------------------------------------------------------
extern "C" void kernel_launch(void* const* d_in, const int* in_sizes, int n_in,
                              void* d_out, int out_size, void* d_ws, size_t ws_size,
                              hipStream_t stream) {
    const float* emb = (const float*)d_in[0];
    const int*   ig  = (const int*)  d_in[1];
    const int*   id  = (const int*)  d_in[2];
    const float* W1  = (const float*)d_in[3];
    const float* b1  = (const float*)d_in[4];
    const float* W2  = (const float*)d_in[5];
    const float* b2  = (const float*)d_in[6];
    float*       out = (float*)d_out;

    const int P = in_sizes[1];             // number of pairs

    // d_ws: 64*8*32 fragments * 32 B = 512 KB of f16 W1 fragments.
    v16h* B = (v16h*)d_ws;
    const int prepThreads = NT_N * KSTEPS * 32;   // 16384
    prep_w1_kernel<<<prepThreads / 256, 256, 0, stream>>>(W1, B);

    const int tiles  = (P + 15) / 16;
    const int blocks = (tiles + 7) / 8;           // 8 waves (tiles) per block
    pair_mlp_kernel<<<blocks, 256, 0, stream>>>(emb, ig, id, B, b1, W2, b2,
                                                out, P);
}
